// MonotonicScalarGatedChannelStack_13073880449228
// MI455X (gfx1250) — compile-verified, matrix operand order checked by reading, not run on hardware
//
#include <hip/hip_runtime.h>
#include <hip/hip_bf16.h>

typedef __attribute__((ext_vector_type(16))) __bf16 v16bf;
typedef __attribute__((ext_vector_type(8)))  __bf16 v8bf;
typedef __attribute__((ext_vector_type(4)))  __bf16 v4bf;
typedef __attribute__((ext_vector_type(8)))  float  v8f;

#define BATCH 4096
#define DIM   2048
#define NCOMP 8
#define CCH   1024
#define NOUT  (NCOMP * CCH)   // 8192
#define TM 128
#define TN 128
#define TK 32
#define LDT 40                 // padded LDS row stride (elements): 80B, 16B-aligned, bank-conflict-free

// ---------------- Gate kernel: g = 4*(1+tanh(x.Wg + b)); G = prefix mask / count -------------
__global__ __launch_bounds__(256) void msgcs_gate(const float* __restrict__ x,
                                                  const float* __restrict__ wg,
                                                  const float* __restrict__ wgb,
                                                  float* __restrict__ Gws,
                                                  float* __restrict__ Gout) {
    const int wid  = threadIdx.x >> 5;
    const int lane = threadIdx.x & 31;
    const int row  = blockIdx.x * 8 + wid;
    const float* xr = x + (size_t)row * DIM;
    float sum = 0.f;
#pragma unroll
    for (int i = 0; i < DIM / (32 * 4); ++i) {
        const int d = (i * 32 + lane) * 4;
        float4 xv = *(const float4*)(xr + d);
        float4 wv = *(const float4*)(wg + d);
        sum += xv.x * wv.x + xv.y * wv.y + xv.z * wv.z + xv.w * wv.w;
    }
#pragma unroll
    for (int off = 16; off; off >>= 1) sum += __shfl_xor(sum, off, 32);

    const float g   = 0.5f * (float)NCOMP * (1.0f + tanhf(sum + wgb[0]));
    const float k   = floorf(g);
    const int   cnt = min((int)k + 1, NCOMP);
    const float inv = 1.0f / (float)cnt;
    if (lane < NCOMP) {
        const float gv = ((float)lane <= k) ? inv : 0.0f;
        Gws[row * NCOMP + lane]  = gv;
        Gout[row * NCOMP + lane] = gv;
    }
}

// --------------- helpers ---------------
__device__ __forceinline__ v16bf ldfrag(const __bf16* p) {
    v8bf a = *(const v8bf*)p;         // K 0..7   (lanes 0-15)  / 8..15  (lanes 16-31)
    v8bf b = *(const v8bf*)(p + 16);  // K 16..23 (lanes 0-15)  / 24..31 (lanes 16-31)
    return __builtin_shufflevector(a, b, 0, 1, 2, 3, 4, 5, 6, 7, 8, 9, 10, 11, 12, 13, 14, 15);
}

__device__ __forceinline__ void cvt4(float4 f, __bf16* hp, __bf16* lp) {
    float fv[4] = {f.x, f.y, f.z, f.w};
    v4bf h, l;
#pragma unroll
    for (int i = 0; i < 4; ++i) {
        __bf16 hh = (__bf16)fv[i];
        h[i] = hh;
        l[i] = (__bf16)(fv[i] - (float)hh);
    }
    *(v4bf*)hp = h;
    *(v4bf*)lp = l;
}

#define WMMA_BF16(A, B, C) \
    __builtin_amdgcn_wmma_f32_16x16x32_bf16(false, (A), false, (B), (short)0, (C), false, false)

// --------------- GEMM:  E[b, n*C+c] = G[b,n] * (x @ W^T + bias)  via bf16x3 split ---------------
__global__ __launch_bounds__(256) void msgcs_gemm(const float* __restrict__ x,
                                                  const float* __restrict__ Wf,   // [8192, 2048]
                                                  const float* __restrict__ bias, // [8192]
                                                  const float* __restrict__ Gws,  // [B, 8]
                                                  float* __restrict__ E) {
    __shared__ __attribute__((aligned(16))) __bf16 Ah[TM * LDT];
    __shared__ __attribute__((aligned(16))) __bf16 Al[TM * LDT];
    __shared__ __attribute__((aligned(16))) __bf16 Bh[TN * LDT];
    __shared__ __attribute__((aligned(16))) __bf16 Bl[TN * LDT];
    __shared__ __attribute__((aligned(16))) float  Gt[TM * NCOMP];

    const int tid  = threadIdx.x;
    const int lane = tid & 31;
    const int wid  = tid >> 5;
    const int half = lane >> 4;   // lane group: K phase
    const int r16  = lane & 15;   // row (A) / col (B) within 16
    const int wm   = wid & 1;     // 2 row groups of 64
    const int wn   = wid >> 1;    // 4 col groups of 32
    const int tileM = blockIdx.y * TM;
    const int tileN = blockIdx.x * TN;

    // stage the per-tile gate scales (contiguous: G is [B,8] row-major)
    {
        const int i = tid * 4;
        *(float4*)(Gt + i) = *(const float4*)(Gws + (size_t)tileM * NCOMP + i);
    }

    const int lr = tid >> 3;        // 0..31  (row within a 32-row slab)
    const int lc = (tid & 7) * 4;   // 0..28  (K column, float4 granular)
    const float* ag = x  + (size_t)(tileM + lr) * DIM + lc;
    const float* bg = Wf + (size_t)(tileN + lr) * DIM + lc;

    float4 aR[4], bR[4];
#pragma unroll
    for (int i = 0; i < 4; ++i) {
        aR[i] = *(const float4*)(ag + (size_t)i * 32 * DIM);
        bR[i] = *(const float4*)(bg + (size_t)i * 32 * DIM);
    }

    v8f acc[4][2];
#pragma unroll
    for (int ms = 0; ms < 4; ++ms)
#pragma unroll
        for (int ns = 0; ns < 2; ++ns) acc[ms][ns] = (v8f){0.f, 0.f, 0.f, 0.f, 0.f, 0.f, 0.f, 0.f};

    const int KSTEPS = DIM / TK;   // 64
    for (int kt = 0; kt < KSTEPS; ++kt) {
        // convert current staged regs to bf16 hi/lo and store to LDS
#pragma unroll
        for (int i = 0; i < 4; ++i) {
            const int e = (lr + i * 32) * LDT + lc;
            cvt4(aR[i], Ah + e, Al + e);
            cvt4(bR[i], Bh + e, Bl + e);
        }
        __syncthreads();

        // software pipeline: issue next K-slab global loads
        if (kt < KSTEPS - 1) {
            const int ko = (kt + 1) * TK;
#pragma unroll
            for (int i = 0; i < 4; ++i) {
                aR[i] = *(const float4*)(ag + ko + (size_t)i * 32 * DIM);
                bR[i] = *(const float4*)(bg + ko + (size_t)i * 32 * DIM);
            }
        }

        // load fragments (two 16B ds reads each, conflict-free padded stride)
        v16bf ah[4], al[4], bh[2], bl[2];
#pragma unroll
        for (int ms = 0; ms < 4; ++ms) {
            const int off = (wm * 64 + ms * 16 + r16) * LDT + half * 8;
            ah[ms] = ldfrag(Ah + off);
            al[ms] = ldfrag(Al + off);
        }
#pragma unroll
        for (int ns = 0; ns < 2; ++ns) {
            const int off = (wn * 32 + ns * 16 + r16) * LDT + half * 8;
            bh[ns] = ldfrag(Bh + off);
            bl[ns] = ldfrag(Bl + off);
        }

        // bf16x3 split: hi*hi + hi*lo + lo*hi  (~f32 accuracy)
#pragma unroll
        for (int ms = 0; ms < 4; ++ms)
#pragma unroll
            for (int ns = 0; ns < 2; ++ns) {
                acc[ms][ns] = WMMA_BF16(ah[ms], bh[ns], acc[ms][ns]);
                acc[ms][ns] = WMMA_BF16(ah[ms], bl[ns], acc[ms][ns]);
                acc[ms][ns] = WMMA_BF16(al[ms], bh[ns], acc[ms][ns]);
            }
        __syncthreads();
    }

    // epilogue: E = G * (acc + bias)
#pragma unroll
    for (int ms = 0; ms < 4; ++ms) {
#pragma unroll
        for (int ns = 0; ns < 2; ++ns) {
            const int gcol = tileN + wn * 32 + ns * 16 + r16;
            const float bv = bias[gcol];
            const int comp = gcol >> 10;   // / CCH
#pragma unroll
            for (int r = 0; r < 8; ++r) {
                const int lrow = wm * 64 + ms * 16 + half * 8 + r;
                const float sc = Gt[lrow * NCOMP + comp];
                E[(size_t)(tileM + lrow) * NOUT + gcol] = (acc[ms][ns][r] + bv) * sc;
            }
        }
    }
}

extern "C" void kernel_launch(void* const* d_in, const int* in_sizes, int n_in,
                              void* d_out, int out_size, void* d_ws, size_t ws_size,
                              hipStream_t stream) {
    (void)in_sizes; (void)n_in; (void)out_size; (void)ws_size;
    const float* x   = (const float*)d_in[0];   // [B, D]
    const float* W   = (const float*)d_in[1];   // [N, C, D] -> [8192, 2048]
    const float* b   = (const float*)d_in[2];   // [N, C]    -> [8192]
    const float* wg  = (const float*)d_in[3];   // [D, 1]
    const float* wgb = (const float*)d_in[4];   // [1]

    float* E    = (float*)d_out;                         // [B, N*C]
    float* Gout = E + (size_t)BATCH * NOUT;              // [B, N]
    float* Gws  = (float*)d_ws;                          // [B, N] scratch

    msgcs_gate<<<BATCH / 8, 256, 0, stream>>>(x, wg, wgb, Gws, Gout);
    msgcs_gemm<<<dim3(NOUT / TN, BATCH / TM), 256, 0, stream>>>(x, W, b, Gws, E);
}
